// LiCompute_41798621724788
// MI455X (gfx1250) — matrix operand order repaired and checked by reading, not compiled, not run on hardware
//
#include <hip/hip_runtime.h>
#include <stdint.h>

typedef __attribute__((ext_vector_type(16))) __bf16 v16bf;
typedef __attribute__((ext_vector_type(8)))  float  v8f;
typedef __attribute__((ext_vector_type(4)))  unsigned int u32x4;
typedef __attribute__((ext_vector_type(8)))  int  i32x8;
typedef __attribute__((ext_vector_type(4)))  int  i32x4;

#define NUM_B  2
#define SEQ_S  4096
#define HEADS  16
#define DIM_D  64
#define KV_T   1024
#define K_SEL  256
#define NEG_MIN (-3.4028234663852886e38f)

// native f32 -> bf16 (backend emits v_cvt_pk_bf16_f32 when pairs fuse)
__device__ __forceinline__ unsigned short f32_to_bf16(float x) {
  __bf16 h = (__bf16)x;
  return __builtin_bit_cast(unsigned short, h);
}

union FragBF {
  v16bf v;
  unsigned short h[16];
  unsigned int   w[8];
};

__global__ __launch_bounds__(256)
void indexer_score_topk(const float* __restrict__ qg,
                        const float* __restrict__ kg,
                        const float* __restrict__ wg,
                        const int*   __restrict__ off_p,
                        int*   __restrict__ idx_out,
                        float* __restrict__ val_out)
{
  // 16 queries x 1024 candidates, f32 scores (also TDM staging for the q tile)
  __shared__ float scores[16 * KV_T];                       // 64 KB
  __shared__ union {
    unsigned short     qtile[16 * HEADS * DIM_D];           // 32 KB (phase 1)
    unsigned long long keys[16 * K_SEL];                    // 32 KB (phase 2)
  } ovl;
  __shared__ float wtile[16 * HEADS];                       // 1 KB
  __shared__ int   hist[8][256];                            // 8 KB
  __shared__ int   wsc[8][4];                               // per-wave scalars

  const int tid  = threadIdx.x;
  const int lane = tid & 31;
  const int wid  = tid >> 5;
  const int hi   = lane >> 4;    // lane group (0 or 1)
  const int lo   = lane & 15;

  const int wgid   = blockIdx.x;
  const int bidx   = wgid / (SEQ_S / 16);
  const int stile  = wgid % (SEQ_S / 16);
  const int s_base = stile * 16;
  const int offset = off_p[0];

  // ---------------- Phase 0: stage q tile (f32 -> bf16) and weights ----------------
  const float* qsrc = qg + ((size_t)bidx * SEQ_S + s_base) * HEADS * DIM_D;
#if __has_builtin(__builtin_amdgcn_tensor_load_to_lds)
  // TDM: DMA the contiguous 64KB q tile (16384 f32) into the scores region,
  // then convert once LDS->LDS into bf16. One wave issues the descriptor.
  if (wid == 0) {
    const unsigned long long ga = (unsigned long long)(uintptr_t)qsrc;
    const unsigned int lds_off  = (unsigned int)(uintptr_t)&scores[0];
    u32x4 g0;
    g0[0] = 1u;                                        // count=1, user mode
    g0[1] = lds_off;                                   // lds_addr
    g0[2] = (unsigned int)(ga & 0xFFFFFFFFull);        // global_addr[31:0]
    g0[3] = ((unsigned int)((ga >> 32) & 0x1FFFFFFull))// global_addr[56:32]
            | (2u << 30);                              // type = 2 ("image")
    i32x8 g1;
    g1[0] = 0x20000;      // data_size = 2 (4 bytes); no multicast/pad/iterate
    g1[1] = 0x40000000;   // tensor_dim0[15:0] = 16384  (<<16)
    g1[2] = 0x00010000;   // tensor_dim1 = 1
    g1[3] = 0x40000000;   // tile_dim0 = 16384 (<<16)
    g1[4] = 1;            // tile_dim1 = 1, tile_dim2 = 0
    g1[5] = 16384;        // tensor_dim0_stride[31:0]
    g1[6] = 0x40000000;   // tensor_dim1_stride[15:0] = 16384 (<<16)
    g1[7] = 0;
    i32x4 g2 = {0, 0, 0, 0};
    i32x4 g3 = {0, 0, 0, 0};
    i32x8 g4 = {0, 0, 0, 0, 0, 0, 0, 0};   // 6-arg toolchain: extra group (unused)
    __builtin_amdgcn_tensor_load_to_lds(g0, g1, g2, g3, g4, 0);
    __builtin_amdgcn_s_wait_tensorcnt(0);
  }
  __syncthreads();
  for (int i = tid; i < 16 * HEADS * DIM_D; i += 256)
    ovl.qtile[i] = f32_to_bf16(scores[i]);
#else
  for (int i = tid; i < 16 * HEADS * DIM_D; i += 256)
    ovl.qtile[i] = f32_to_bf16(qsrc[i]);
#endif
  {
    const float* wsrc = wg + ((size_t)bidx * SEQ_S + s_base) * HEADS;
    for (int i = tid; i < 16 * HEADS; i += 256)
      wtile[i] = wsrc[i];
  }
  __syncthreads();

  // ---------------- Phase 1: WMMA score panel [16 x 1024] ----------------
  // Each wave owns 8 t-tiles: tt = wid + 8*i. B fragments and accumulators
  // stay register-resident; A fragments are loaded once per head.
  {
    const float* kbase = kg + (size_t)bidx * KV_T * DIM_D;

    FragBF bfr[8][2];          // k-side fragments, all 8 tiles x 2 K-blocks
    #pragma unroll
    for (int i = 0; i < 8; ++i) {
      const int t0 = (wid + 8 * i) * 16;
      #pragma unroll
      for (int kb = 0; kb < 2; ++kb) {
        // B layout: half j<8 -> K = j + 8*hi, j>=8 -> K = 16 + 8*hi + (j-8), N = lo
        const float* krow = kbase + (size_t)(t0 + lo) * DIM_D + kb * 32 + hi * 8;
        #pragma unroll
        for (int j = 0; j < 16; ++j) bfr[i][kb].h[j] = f32_to_bf16(krow[j + (j & 8)]);
      }
    }

    v8f accs[8];
    #pragma unroll
    for (int i = 0; i < 8; ++i) accs[i] = (v8f){0.f,0.f,0.f,0.f,0.f,0.f,0.f,0.f};

    for (int hh = 0; hh < HEADS; ++hh) {
      // A fragments (q side): row = lo; halves 0..7 -> dims base..+7,
      // halves 8..15 -> dims base+16..+23  (base = kb*32 + 8*hi)
      FragBF a0, a1;
      {
        const int b0 = (lo * HEADS + hh) * DIM_D + hi * 8;
        #pragma unroll
        for (int j = 0; j < 8; ++j) {
          a0.h[j]     = ovl.qtile[b0 + j];
          a0.h[8 + j] = ovl.qtile[b0 + 16 + j];
          a1.h[j]     = ovl.qtile[b0 + 32 + j];
          a1.h[8 + j] = ovl.qtile[b0 + 48 + j];
        }
      }
      float wv[8];
      #pragma unroll
      for (int v = 0; v < 8; ++v) wv[v] = wtile[(v + hi * 8) * HEADS + hh];

      #pragma unroll
      for (int i = 0; i < 8; ++i) {
        v8f c = {0.f,0.f,0.f,0.f,0.f,0.f,0.f,0.f};
        c = __builtin_amdgcn_wmma_f32_16x16x32_bf16(
                false, a0.v, false, bfr[i][0].v, (short)0, c, false, false);
        c = __builtin_amdgcn_wmma_f32_16x16x32_bf16(
                false, a1.v, false, bfr[i][1].v, (short)0, c, false, false);
        #pragma unroll
        for (int v = 0; v < 8; ++v) {
          float cv = c[v];
          cv = cv > 0.f ? cv : 0.f;
          accs[i][v] += cv * wv[v];
        }
      }
    }

    // causal mask (+NEG like reference) and store the score panel
    #pragma unroll
    for (int i = 0; i < 8; ++i) {
      const int t = (wid + 8 * i) * 16 + lo;
      #pragma unroll
      for (int v = 0; v < 8; ++v) {
        const int m  = v + hi * 8;
        const int sg = s_base + m;
        float val = accs[i][v];
        if (t >= ((sg + 1) >> 2)) val += NEG_MIN;
        scores[m * KV_T + t] = val;
      }
    }
  }
  __syncthreads();

  // ---------------- Phase 2: per-query top-256 (radix select + bitonic) ----------------
  for (int qi = 0; qi < 2; ++qi) {
    const int qr = wid * 2 + qi;                 // query row in tile
    const float* srow = &scores[qr * KV_T];

    // monotone uint keys; lane owns t = 32*i + lane
    unsigned int k32[32];
    #pragma unroll
    for (int i = 0; i < 32; ++i) {
      unsigned int bu = __builtin_bit_cast(unsigned int, srow[i * 32 + lane]);
      k32[i] = (bu & 0x80000000u) ? ~bu : (bu | 0x80000000u);
    }

    // 4 x 8-bit radix passes to find the K_SEL-th largest key
    unsigned int prefix = 0, pmask = 0;
    int krem = K_SEL;
    for (int pass = 0; pass < 4; ++pass) {
      const int shift = 24 - 8 * pass;
      for (int bnum = lane; bnum < 256; bnum += 32) hist[wid][bnum] = 0;
      __syncthreads();
      #pragma unroll
      for (int i = 0; i < 32; ++i) {
        const unsigned int kk = k32[i];
        if ((kk & pmask) == prefix) atomicAdd(&hist[wid][(kk >> shift) & 0xFF], 1);
      }
      __syncthreads();
      if (lane == 0) {
        int rem = krem, bin = 0;
        for (int bb = 255; bb >= 0; --bb) {
          const int cnt = hist[wid][bb];
          if (cnt >= rem) { bin = bb; break; }
          rem -= cnt;
        }
        wsc[wid][0] = bin;
        wsc[wid][1] = rem;
      }
      __syncthreads();
      prefix |= ((unsigned int)wsc[wid][0]) << shift;
      pmask  |= 0xFFu << shift;
      krem    = wsc[wid][1];
    }
    const unsigned int thr = prefix;             // exact K-th largest key

    // collect: all keys > thr, then 'krem' keys == thr by ascending index
    if (lane == 0) wsc[wid][2] = 0;
    __syncthreads();
    unsigned long long* kq = &ovl.keys[qr * K_SEL];
    #pragma unroll
    for (int i = 0; i < 32; ++i) {
      const unsigned int kk = k32[i];
      if (kk > thr) {
        const int slot = atomicAdd(&wsc[wid][2], 1);
        const unsigned int t = (unsigned int)(i * 32 + lane);
        kq[slot] = ((unsigned long long)kk << 32) | (unsigned int)(~t);
      }
    }
    __syncthreads();
    const int cgt = wsc[wid][2];
    int taken = 0;
    #pragma unroll
    for (int i = 0; i < 32; ++i) {
      const unsigned int kk = k32[i];
      const bool eq = (kk == thr);
      const unsigned int mask = (unsigned int)__ballot(eq);
      if (eq) {
        const int rank = taken + __builtin_popcount(mask & ((1u << lane) - 1u));
        if (rank < krem) {
          const unsigned int t = (unsigned int)(i * 32 + lane);
          kq[cgt + rank] = ((unsigned long long)kk << 32) | (unsigned int)(~t);
        }
      }
      taken += __builtin_popcount(mask);
    }

    // bitonic sort ascending (composite key: value, then ~t => ties pick low t)
    for (int size = 2; size <= K_SEL; size <<= 1) {
      for (int stride = size >> 1; stride > 0; stride >>= 1) {
        __syncthreads();
        #pragma unroll
        for (int r = 0; r < K_SEL / 32; ++r) {
          const int e = lane + 32 * r;
          const int p = e ^ stride;
          if (p > e) {
            const unsigned long long a = kq[e], b = kq[p];
            const bool up = ((e & size) == 0);
            if ((a > b) == up) { kq[e] = b; kq[p] = a; }
          }
        }
      }
    }
    __syncthreads();

    // emit descending; decode keys back to (t, score)
    const int sg   = s_base + qr;
    const int tmax = (sg + 1) >> 2;
    const size_t obase = ((size_t)bidx * SEQ_S + sg) * K_SEL;
    #pragma unroll
    for (int r = 0; r < K_SEL / 32; ++r) {
      const int j = lane + 32 * r;
      const unsigned long long ckey = kq[K_SEL - 1 - j];
      const unsigned int kk = (unsigned int)(ckey >> 32);
      const unsigned int t  = ~((unsigned int)ckey);
      const unsigned int bu = (kk & 0x80000000u) ? (kk & 0x7FFFFFFFu) : ~kk;
      idx_out[obase + j] = ((int)t >= tmax) ? -1 : ((int)t + offset);
      val_out[obase + j] = __builtin_bit_cast(float, bu);
    }
    __syncthreads();
  }
}

extern "C" void kernel_launch(void* const* d_in, const int* in_sizes, int n_in,
                              void* d_out, int out_size, void* d_ws, size_t ws_size,
                              hipStream_t stream) {
  (void)in_sizes; (void)n_in; (void)out_size; (void)d_ws; (void)ws_size;
  const float* q   = (const float*)d_in[0];
  const float* k   = (const float*)d_in[1];
  const float* w   = (const float*)d_in[2];
  const int*   off = (const int*)d_in[4];      // d_in[3] = seqlen (compile-time 4096)

  int*   idx_out = (int*)d_out;
  float* val_out = (float*)d_out + (size_t)NUM_B * SEQ_S * K_SEL;

  const int blocks = NUM_B * (SEQ_S / 16);     // 512 workgroups of 16 queries
  indexer_score_topk<<<blocks, 256, 0, stream>>>(q, k, w, off, idx_out, val_out);
}
